// NozzleGNN_42949672960307
// MI455X (gfx1250) — matrix operand order repaired and checked by reading, not compiled, not run on hardware
//
#include <hip/hip_runtime.h>
#include <hip/hip_bf16.h>

typedef __bf16 v16bf __attribute__((ext_vector_type(16)));
typedef float  v8f   __attribute__((ext_vector_type(8)));
typedef unsigned int uiv4 __attribute__((ext_vector_type(4)));

#define HID 64
#define NK 128          // node-MLP K
#define WPB 4           // waves per block for WMMA kernels

__device__ __forceinline__ float silu_f(float x) {
  return x * __builtin_amdgcn_rcpf(1.0f + __expf(-x));
}

// ---- WMMA fragment loaders (bf16, 16x16x32) -------------------------------
// A-matrix 16x32 bf16 layout: lanes 0-15 -> M=lane, K {0..7}U{16..23};
// lanes 16-31 -> M=lane-16, K {8..15}U{24..31}.
// Generic 2x16B gather: elements 0..7 = p[0..7], 8..15 = p[16..23].
__device__ __forceinline__ v16bf frag_2x16(const __bf16* p) {
  union { uiv4 q[2]; v16bf v; } u;
  u.q[0] = *(const uiv4*)(p);
  u.q[1] = *(const uiv4*)(p + 16);
  return u.v;
}
// A-fragment from row-major LDS.
__device__ __forceinline__ v16bf ldsA16(const __bf16* base, int strideK, int lane, int kbase) {
  int m  = lane & 15;
  int k0 = kbase + ((lane & 16) ? 8 : 0);
  return frag_2x16(base + m * strideK + k0);
}
// B-matrix 32x16 bf16: lanes 0-15 -> N=lane, K {0..7}U{16..23}; lanes 16-31 -> N=lane-16,
// K {8..15}U{24..31}. Weights stored transposed in LDS as Wt[N][K] so reads are contiguous.
__device__ __forceinline__ v16bf ldsB16(const __bf16* wt, int strideK, int lane, int ntile, int kbase) {
  int n  = (ntile << 4) + (lane & 15);
  int k0 = kbase + ((lane & 16) ? 8 : 0);
  const __bf16* p = wt + n * strideK + k0;
  union { uiv4 q[2]; v16bf v; } u;
  u.q[0] = *(const uiv4*)(p);
  u.q[1] = *(const uiv4*)(p + 16);
  return u.v;
}

// ---- Encoder: h = LN(silu([x|gv] @ W1 + b1) @ W2 + b2) --------------------
__global__ __launch_bounds__(64) void encoder_kernel(
    const float* __restrict__ x, const int* __restrict__ batch,
    const float* __restrict__ cp, const float* __restrict__ bp,
    const float* __restrict__ W1, const float* __restrict__ b1,
    const float* __restrict__ W2, const float* __restrict__ b2,
    const float* __restrict__ g, const float* __restrict__ be,
    float* __restrict__ h, __bf16* __restrict__ hb) {
  __shared__ float in16[16];
  __shared__ float a1[64];
  __shared__ float h2[64];
  __shared__ float stats[2];
  int node = blockIdx.x, t = threadIdx.x;
  if (t < 16) {
    int b = batch[node];
    float v;
    if (t < 8)       v = x[node * 8 + t];
    else if (t < 12) v = cp[b * 4 + t - 8];
    else             v = bp[b * 4 + t - 12];
    in16[t] = v;
  }
  __syncthreads();
  float s = b1[t];
#pragma unroll
  for (int k = 0; k < 16; ++k) s += in16[k] * W1[k * 64 + t];
  a1[t] = silu_f(s);
  __syncthreads();
  float s2 = b2[t];
  for (int k = 0; k < 64; ++k) s2 += a1[k] * W2[k * 64 + t];
  h2[t] = s2;
  __syncthreads();
  if (t == 0) {
    float m = 0.f;
    for (int k = 0; k < 64; ++k) m += h2[k];
    m *= (1.f / 64.f);
    float v = 0.f;
    for (int k = 0; k < 64; ++k) { float d = h2[k] - m; v += d * d; }
    stats[0] = m;
    stats[1] = rsqrtf(v * (1.f / 64.f) + 1e-5f);
  }
  __syncthreads();
  float val = (s2 - stats[0]) * stats[1] * g[t] + be[t];
  h[(size_t)node * 64 + t]  = val;
  hb[(size_t)node * 64 + t] = (__bf16)val;
}

// ---- Edge MLP + scatter-add (WMMA, reg weights, direct global A-frags) ----
__global__ __launch_bounds__(128) void edge_mlp_kernel(
    const __bf16* __restrict__ hb, const int* __restrict__ ei,
    const float* __restrict__ ea,
    const float* __restrict__ W1, const float* __restrict__ b1,
    const float* __restrict__ W2, const float* __restrict__ b2,
    const float* __restrict__ g, const float* __restrict__ be,
    float* __restrict__ agg, int E, int numTiles) {
  // Phase 1: Wt1 [64][160] bf16 (20480) + Wt2 [64][64] bf16 (8192) = 28672
  // Phase 2 (same region): per wave H[16][64] bf16 (2048) + O[16][64] f32
  //   (4096) = 6144 * WPB = 24576 <= 28672
  __shared__ __align__(16) unsigned char smem[28672];
  __shared__ float sBias[256];   // b1 | b2 | g | be
  int lane = threadIdx.x & 31;
  int wv = threadIdx.x >> 5;

  // ---- stage weights transposed + bias vectors ----------------------------
  {
    __bf16* sW1 = (__bf16*)smem;
    __bf16* sW2 = (__bf16*)(smem + 20480);
    for (int i = threadIdx.x; i < 64 * 160; i += blockDim.x) {
      int n = i / 160, k = i % 160;
      sW1[i] = (k < 133) ? (__bf16)W1[k * 64 + n] : (__bf16)0.f;
    }
    for (int i = threadIdx.x; i < 64 * 64; i += blockDim.x) {
      int n = i >> 6, k = i & 63;
      sW2[i] = (__bf16)W2[k * 64 + n];
    }
    for (int i = threadIdx.x; i < 64; i += blockDim.x) {
      sBias[i]       = b1[i];
      sBias[64 + i]  = b2[i];
      sBias[128 + i] = g[i];
      sBias[192 + i] = be[i];
    }
  }
  __syncthreads();
  v16bf w1f[20];   // [kc][t] -> kc*4+t
  v16bf w2f[8];
  {
    const __bf16* sW1 = (const __bf16*)smem;
    const __bf16* sW2 = (const __bf16*)(smem + 20480);
#pragma unroll
    for (int kc = 0; kc < 5; ++kc)
#pragma unroll
      for (int t = 0; t < 4; ++t) w1f[kc * 4 + t] = ldsB16(sW1, 160, lane, t, kc * 32);
#pragma unroll
    for (int kc = 0; kc < 2; ++kc)
#pragma unroll
      for (int t = 0; t < 4; ++t) w2f[kc * 4 + t] = ldsB16(sW2, 64, lane, t, kc * 32);
  }
  __syncthreads();   // weight LDS dead; region reused as per-wave scratch

  unsigned char* pw = smem + (size_t)wv * 6144;
  __bf16* sH = (__bf16*)pw;            // [16][64] bf16
  float*  sO = (float*)(pw + 2048);    // [16][64] f32

  int row = lane & 15, half = lane >> 4;
  int koffA = half ? 8 : 0;            // A-layout sub-chunk offset
  int wid = blockIdx.x * WPB + wv;
  int wstride = gridDim.x * WPB;

  for (int tile = wid; tile < numTiles; tile += wstride) {
    int e = tile * 16 + row;
    bool valid = e < E;
    int dstIdx = valid ? ei[E + e] : 0;
    int srcIdx = valid ? ei[e] : 0;
    // ---- A fragments straight from global: K0..63 = h[dst], 64..127 = h[src]
    v16bf af[4];
    if (valid) {
      const __bf16* pd = hb + (size_t)dstIdx * 64 + koffA;
      const __bf16* ps = hb + (size_t)srcIdx * 64 + koffA;
      af[0] = frag_2x16(pd);
      af[1] = frag_2x16(pd + 32);
      af[2] = frag_2x16(ps);
      af[3] = frag_2x16(ps + 32);
    } else {
      union { uiv4 q[2]; v16bf v; } z;
      z.q[0] = (uiv4){0u, 0u, 0u, 0u};
      z.q[1] = (uiv4){0u, 0u, 0u, 0u};
#pragma unroll
      for (int i = 0; i < 4; ++i) af[i] = z.v;
    }
    // K 128..159: edge_attr (5) + zero pad; nonzero only for lanes 0-15
    v16bf a4;
    {
      union { uiv4 q[2]; v16bf v; } u4;
      u4.q[0] = (uiv4){0u, 0u, 0u, 0u};
      u4.q[1] = (uiv4){0u, 0u, 0u, 0u};
      if (valid && !half) {
#pragma unroll
        for (int j = 0; j < 5; ++j) u4.v[j] = (__bf16)ea[(size_t)e * 5 + j];
      }
      a4 = u4.v;
    }
    // ---- GEMM1: [16x160] x [160x64] ---------------------------------------
    v8f acc[4];
#pragma unroll
    for (int t = 0; t < 4; ++t)
#pragma unroll
      for (int r = 0; r < 8; ++r) acc[t][r] = 0.f;
#pragma unroll
    for (int kc = 0; kc < 4; ++kc)
#pragma unroll
      for (int t = 0; t < 4; ++t)
        acc[t] = __builtin_amdgcn_wmma_f32_16x16x32_bf16(false, af[kc], false, w1f[kc * 4 + t],
                                                         (short)0, acc[t], false, false);
#pragma unroll
    for (int t = 0; t < 4; ++t)
      acc[t] = __builtin_amdgcn_wmma_f32_16x16x32_bf16(false, a4, false, w1f[16 + t],
                                                       (short)0, acc[t], false, false);
    // ---- bias + silu -> sH (bf16) -----------------------------------------
    int nl = lane & 15, mb = half * 8;
#pragma unroll
    for (int t = 0; t < 4; ++t) {
      int n = t * 16 + nl;
      float bb = sBias[n];
#pragma unroll
      for (int r = 0; r < 8; ++r)
        sH[(mb + r) * 64 + n] = (__bf16)silu_f(acc[t][r] + bb);
    }
    // ---- GEMM2: [16x64] x [64x64] -----------------------------------------
    v8f a2[4];
#pragma unroll
    for (int t = 0; t < 4; ++t)
#pragma unroll
      for (int r = 0; r < 8; ++r) a2[t][r] = 0.f;
#pragma unroll
    for (int kc = 0; kc < 2; ++kc) {
      v16bf a = ldsA16(sH, 64, lane, kc * 32);
#pragma unroll
      for (int t = 0; t < 4; ++t)
        a2[t] = __builtin_amdgcn_wmma_f32_16x16x32_bf16(false, a, false, w2f[kc * 4 + t],
                                                        (short)0, a2[t], false, false);
    }
    // ---- +bias -> sO (f32) ------------------------------------------------
#pragma unroll
    for (int t = 0; t < 4; ++t) {
      int n = t * 16 + nl;
      float bb = sBias[64 + n];
#pragma unroll
      for (int r = 0; r < 8; ++r) sO[(mb + r) * 64 + n] = a2[t][r] + bb;
    }
    // ---- LayerNorm per edge + scatter-add; all 32 lanes participate -------
    if (valid) {
      int coff = half * 32;
      const float* rp = sO + row * 64 + coff;
      float vals[32];
#pragma unroll
      for (int i = 0; i < 32; ++i) vals[i] = rp[i];
      float ps = 0.f;
#pragma unroll
      for (int i = 0; i < 32; ++i) ps += vals[i];
      float m = (ps + __shfl_xor(ps, 16)) * (1.f / 64.f);
      float pv = 0.f;
#pragma unroll
      for (int i = 0; i < 32; ++i) { float d = vals[i] - m; pv += d * d; }
      float var = (pv + __shfl_xor(pv, 16)) * (1.f / 64.f);
      float rs = rsqrtf(var + 1e-5f);
      float* ap = agg + (size_t)dstIdx * 64 + coff;
#pragma unroll
      for (int i = 0; i < 32; ++i)
        atomicAdd(ap + i, (vals[i] - m) * rs * sBias[128 + coff + i] + sBias[192 + coff + i]);
    }
  }
}

// ---- Node MLP + residual (WMMA, reg weights, direct global A-frags) -------
__global__ __launch_bounds__(128) void node_mlp_kernel(
    float* __restrict__ h, __bf16* __restrict__ hb,
    const float* __restrict__ agg,
    const float* __restrict__ W1, const float* __restrict__ b1,
    const float* __restrict__ W2, const float* __restrict__ b2,
    const float* __restrict__ g, const float* __restrict__ be,
    int N, int numTiles) {
  // Phase 1: Wt1 [64][128] (16384) + Wt2 [64][64] (8192) = 24576
  // Phase 2 (same region): per wave H (2048) + O (4096) = 6144 * WPB = 24576
  __shared__ __align__(16) unsigned char smem[24576];
  __shared__ float sBias[256];
  int lane = threadIdx.x & 31;
  int wv = threadIdx.x >> 5;

  {
    __bf16* sW1 = (__bf16*)smem;
    __bf16* sW2 = (__bf16*)(smem + 16384);
    for (int i = threadIdx.x; i < 64 * NK; i += blockDim.x) {
      int n = i / NK, k = i % NK;
      sW1[i] = (__bf16)W1[k * 64 + n];
    }
    for (int i = threadIdx.x; i < 64 * 64; i += blockDim.x) {
      int n = i >> 6, k = i & 63;
      sW2[i] = (__bf16)W2[k * 64 + n];
    }
    for (int i = threadIdx.x; i < 64; i += blockDim.x) {
      sBias[i]       = b1[i];
      sBias[64 + i]  = b2[i];
      sBias[128 + i] = g[i];
      sBias[192 + i] = be[i];
    }
  }
  __syncthreads();
  v16bf w1f[16];
  v16bf w2f[8];
  {
    const __bf16* sW1 = (const __bf16*)smem;
    const __bf16* sW2 = (const __bf16*)(smem + 16384);
#pragma unroll
    for (int kc = 0; kc < 4; ++kc)
#pragma unroll
      for (int t = 0; t < 4; ++t) w1f[kc * 4 + t] = ldsB16(sW1, NK, lane, t, kc * 32);
#pragma unroll
    for (int kc = 0; kc < 2; ++kc)
#pragma unroll
      for (int t = 0; t < 4; ++t) w2f[kc * 4 + t] = ldsB16(sW2, 64, lane, t, kc * 32);
  }
  __syncthreads();

  unsigned char* pw = smem + (size_t)wv * 6144;
  __bf16* sH = (__bf16*)pw;
  float*  sO = (float*)(pw + 2048);

  int row = lane & 15, half = lane >> 4;
  int koffA = half ? 8 : 0;
  int wid = blockIdx.x * WPB + wv;
  int wstride = gridDim.x * WPB;

  for (int tile = wid; tile < numTiles; tile += wstride) {
    int node = tile * 16 + row;
    bool valid = node < N;
    // ---- A fragments: K0..63 = h (bf16 direct), K64..127 = agg (f32->bf16)
    v16bf af[4];
    if (valid) {
      const __bf16* ph = hb + (size_t)node * 64 + koffA;
      af[0] = frag_2x16(ph);
      af[1] = frag_2x16(ph + 32);
      const float* pa = agg + (size_t)node * 64 + koffA;
#pragma unroll
      for (int c = 0; c < 2; ++c) {
        union { uiv4 q[2]; v16bf v; } u;
        u.q[0] = (uiv4){0u, 0u, 0u, 0u};
        u.q[1] = (uiv4){0u, 0u, 0u, 0u};
        const float* p = pa + c * 32;
#pragma unroll
        for (int j = 0; j < 8; ++j) {
          u.v[j]     = (__bf16)p[j];
          u.v[8 + j] = (__bf16)p[16 + j];
        }
        af[2 + c] = u.v;
      }
    } else {
      union { uiv4 q[2]; v16bf v; } z;
      z.q[0] = (uiv4){0u, 0u, 0u, 0u};
      z.q[1] = (uiv4){0u, 0u, 0u, 0u};
#pragma unroll
      for (int i = 0; i < 4; ++i) af[i] = z.v;
    }
    // ---- GEMM1: [16x128] x [128x64] ---------------------------------------
    v8f acc[4];
#pragma unroll
    for (int t = 0; t < 4; ++t)
#pragma unroll
      for (int r = 0; r < 8; ++r) acc[t][r] = 0.f;
#pragma unroll
    for (int kc = 0; kc < 4; ++kc)
#pragma unroll
      for (int t = 0; t < 4; ++t)
        acc[t] = __builtin_amdgcn_wmma_f32_16x16x32_bf16(false, af[kc], false, w1f[kc * 4 + t],
                                                         (short)0, acc[t], false, false);
    int nl = lane & 15, mb = half * 8;
#pragma unroll
    for (int t = 0; t < 4; ++t) {
      int n = t * 16 + nl;
      float bb = sBias[n];
#pragma unroll
      for (int r = 0; r < 8; ++r)
        sH[(mb + r) * 64 + n] = (__bf16)silu_f(acc[t][r] + bb);
    }
    // ---- GEMM2 ------------------------------------------------------------
    v8f a2[4];
#pragma unroll
    for (int t = 0; t < 4; ++t)
#pragma unroll
      for (int r = 0; r < 8; ++r) a2[t][r] = 0.f;
#pragma unroll
    for (int kc = 0; kc < 2; ++kc) {
      v16bf a = ldsA16(sH, 64, lane, kc * 32);
#pragma unroll
      for (int t = 0; t < 4; ++t)
        a2[t] = __builtin_amdgcn_wmma_f32_16x16x32_bf16(false, a, false, w2f[kc * 4 + t],
                                                        (short)0, a2[t], false, false);
    }
#pragma unroll
    for (int t = 0; t < 4; ++t) {
      int n = t * 16 + nl;
      float bb = sBias[64 + n];
#pragma unroll
      for (int r = 0; r < 8; ++r) sO[(mb + r) * 64 + n] = a2[t][r] + bb;
    }
    // ---- LayerNorm + residual; lane pair splits the row 32/32 -------------
    if (valid) {
      int coff = half * 32;
      const float* rp = sO + row * 64 + coff;
      float vals[32];
#pragma unroll
      for (int i = 0; i < 32; ++i) vals[i] = rp[i];
      float ps = 0.f;
#pragma unroll
      for (int i = 0; i < 32; ++i) ps += vals[i];
      float m = (ps + __shfl_xor(ps, 16)) * (1.f / 64.f);
      float pv = 0.f;
#pragma unroll
      for (int i = 0; i < 32; ++i) { float d = vals[i] - m; pv += d * d; }
      float var = (pv + __shfl_xor(pv, 16)) * (1.f / 64.f);
      float rs = rsqrtf(var + 1e-5f);
      float* hp = h + (size_t)node * 64 + coff;
      __bf16* hbp = hb + (size_t)node * 64 + coff;
#pragma unroll
      for (int i = 0; i < 32; ++i) {
        float hn = hp[i] + (vals[i] - m) * rs * sBias[128 + coff + i] + sBias[192 + coff + i];
        hp[i] = hn;
        hbp[i] = (__bf16)hn;
      }
    }
  }
}

// ---- Local decoder: out = silu(h@W1+b1)@W2+b2 -> [N,6] --------------------
__global__ __launch_bounds__(64) void decoder_local_kernel(
    const float* __restrict__ h,
    const float* __restrict__ W1, const float* __restrict__ b1,
    const float* __restrict__ W2, const float* __restrict__ b2,
    float* __restrict__ out) {
  __shared__ float hr[64];
  __shared__ float a1[64];
  int node = blockIdx.x, t = threadIdx.x;
  hr[t] = h[(size_t)node * 64 + t];
  __syncthreads();
  float s = b1[t];
  for (int k = 0; k < 64; ++k) s += hr[k] * W1[k * 64 + t];
  a1[t] = silu_f(s);
  __syncthreads();
  if (t < 6) {
    float s2 = b2[t];
    for (int k = 0; k < 64; ++k) s2 += a1[k] * W2[k * 6 + t];
    out[(size_t)node * 6 + t] = s2;
  }
}

// ---- Pooling --------------------------------------------------------------
__global__ void pool_init_kernel(float* pooled, float* cnt, int B) {
  int t = threadIdx.x;
  if (t < B * 64) pooled[t] = 0.f;
  if (t < B) cnt[t] = 0.f;
}
__global__ void pool_accum_kernel(const float* __restrict__ h,
                                  const int* __restrict__ batch,
                                  float* __restrict__ pooled,
                                  float* __restrict__ cnt, int N) {
  long long idx = (long long)blockIdx.x * blockDim.x + threadIdx.x;
  if (idx >= (long long)N * 64) return;
  int node = (int)(idx >> 6), f = (int)(idx & 63);
  int b = batch[node];
  atomicAdd(&pooled[b * 64 + f], h[idx]);
  if (f == 0) atomicAdd(&cnt[b], 1.0f);
}
__global__ __launch_bounds__(128) void decoder_global_kernel(
    const float* __restrict__ pooled, const float* __restrict__ cnt,
    const float* __restrict__ W1, const float* __restrict__ b1,
    const float* __restrict__ W2, const float* __restrict__ b2,
    float* __restrict__ out) {
  __shared__ float pm[4 * 64];
  __shared__ float hd[4 * 32];
  int t = threadIdx.x;
  for (int i = t; i < 256; i += 128) {
    int gg = i >> 6;
    float c = cnt[gg];
    if (c < 1.f) c = 1.f;
    pm[i] = pooled[i] / c;
  }
  __syncthreads();
  {
    int gg = t >> 5, j = t & 31;
    float s = b1[j];
    for (int k = 0; k < 64; ++k) s += pm[gg * 64 + k] * W1[k * 32 + j];
    hd[gg * 32 + j] = silu_f(s);
  }
  __syncthreads();
  if (t < 16) {
    int gg = t >> 2, k = t & 3;
    float s = b2[k];
    for (int i = 0; i < 32; ++i) s += hd[gg * 32 + i] * W2[i * 4 + k];
    out[gg * 4 + k] = s;
  }
}

extern "C" void kernel_launch(void* const* d_in, const int* in_sizes, int n_in,
                              void* d_out, int out_size, void* d_ws, size_t ws_size,
                              hipStream_t stream) {
  const float* x     = (const float*)d_in[0];
  const int*   ei    = (const int*)d_in[1];
  const float* ea    = (const float*)d_in[2];
  const int*   batch = (const int*)d_in[3];
  const float* cp    = (const float*)d_in[4];
  const float* bp    = (const float*)d_in[5];
  const float* encW1 = (const float*)d_in[6];
  const float* encb1 = (const float*)d_in[7];
  const float* encW2 = (const float*)d_in[8];
  const float* encb2 = (const float*)d_in[9];
  const float* encg  = (const float*)d_in[10];
  const float* encbe = (const float*)d_in[11];
  const float* eW1   = (const float*)d_in[12];
  const float* eb1   = (const float*)d_in[13];
  const float* eW2   = (const float*)d_in[14];
  const float* eb2   = (const float*)d_in[15];
  const float* eg    = (const float*)d_in[16];
  const float* ebe   = (const float*)d_in[17];
  const float* nW1   = (const float*)d_in[18];
  const float* nb1   = (const float*)d_in[19];
  const float* nW2   = (const float*)d_in[20];
  const float* nb2   = (const float*)d_in[21];
  const float* ng    = (const float*)d_in[22];
  const float* nbe   = (const float*)d_in[23];
  const float* dlW1  = (const float*)d_in[24];
  const float* dlb1  = (const float*)d_in[25];
  const float* dlW2  = (const float*)d_in[26];
  const float* dlb2  = (const float*)d_in[27];
  const float* dgW1  = (const float*)d_in[28];
  const float* dgb1  = (const float*)d_in[29];
  const float* dgW2  = (const float*)d_in[30];
  const float* dgb2  = (const float*)d_in[31];

  const int N = in_sizes[0] / 8;
  const int E = in_sizes[2] / 5;
  const int L = in_sizes[13] / 64;   // eb1 is [L,64]
  const int B = in_sizes[4] / 4;

  // workspace carve (256B aligned)
  char* ws = (char*)d_ws;
  size_t off = 0;
  auto carve = [&](size_t bytes) {
    void* p = ws + off;
    off = (off + bytes + 255) & ~(size_t)255;
    return p;
  };
  float*  h      = (float*)carve((size_t)N * 64 * sizeof(float));
  __bf16* hbuf   = (__bf16*)carve((size_t)N * 64 * sizeof(__bf16));
  float*  agg    = (float*)carve((size_t)N * 64 * sizeof(float));
  float*  pooled = (float*)carve((size_t)B * 64 * sizeof(float));
  float*  cntv   = (float*)carve((size_t)B * sizeof(float));
  (void)ws_size; (void)n_in; (void)out_size;

  encoder_kernel<<<N, 64, 0, stream>>>(x, batch, cp, bp, encW1, encb1, encW2,
                                       encb2, encg, encbe, h, hbuf);

  const int edgeTiles = (E + 15) / 16;
  const int nodeTiles = (N + 15) / 16;
  for (int l = 0; l < L; ++l) {
    hipMemsetAsync(agg, 0, (size_t)N * 64 * sizeof(float), stream);
    edge_mlp_kernel<<<1200, 128, 0, stream>>>(
        hbuf, ei, ea, eW1 + (size_t)l * 133 * 64, eb1 + l * 64,
        eW2 + (size_t)l * 64 * 64, eb2 + l * 64, eg + l * 64, ebe + l * 64,
        agg, E, edgeTiles);
    node_mlp_kernel<<<480, 128, 0, stream>>>(
        h, hbuf, agg, nW1 + (size_t)l * 128 * 64, nb1 + l * 64,
        nW2 + (size_t)l * 64 * 64, nb2 + l * 64, ng + l * 64, nbe + l * 64,
        N, nodeTiles);
  }

  decoder_local_kernel<<<N, 64, 0, stream>>>(h, dlW1, dlb1, dlW2, dlb2,
                                             (float*)d_out);
  pool_init_kernel<<<1, 320, 0, stream>>>(pooled, cntv, B);
  long long tot = (long long)N * 64;
  pool_accum_kernel<<<(int)((tot + 255) / 256), 256, 0, stream>>>(h, batch, pooled, cntv, N);
  decoder_global_kernel<<<1, 128, 0, stream>>>(pooled, cntv, dgW1, dgb1, dgW2,
                                               dgb2, (float*)d_out + (size_t)N * 6);
}
